// DavidBeansV2_29222957481993
// MI455X (gfx1250) — matrix operand order repaired and checked by reading, not compiled, not run on hardware
//
#include <hip/hip_runtime.h>
#include <hip/hip_bf16.h>

// CDNA5 / gfx1250, wave32. bf16 WMMA 16x16x32 (f32 acc) + async LDS copies.

typedef __attribute__((ext_vector_type(16))) __bf16 v16bf;
typedef __attribute__((ext_vector_type(8)))  __bf16 v8bf;
typedef __attribute__((ext_vector_type(8)))  float  v8f;
typedef __attribute__((ext_vector_type(4)))  int    v4i;

#define B_   8
#define P_   1024
#define D_   768
#define K_   16
#define NEGV -1000000000.0f
#define DPAD (D_ + 8)   // LDS row pad: 1552B stride -> conflict-free column reads

#if defined(__has_builtin)
#if __has_builtin(__builtin_amdgcn_global_load_async_to_lds_b128)
#define HAVE_ASYNC 1
#endif
#endif

// async b128 copy operands: (global v4i*, lds v4i*, imm offset, imm cpol)
#define GLB_V4(p) ((__attribute__((address_space(1))) v4i*)(p))
#define LDS_V4(p) ((__attribute__((address_space(3))) v4i*)(p))

// ---------------------------------------------------------------------------
// Kernel 1: projections  out = xp @ W + b  for W in {Wq, Wk, Wv}
// block = 256 threads = 8 waves; each wave computes one 16x16 output tile.
// Shared 768x16 W column-panel staged in LDS as bf16 (reused by all 8 waves).
// grid = (48 Ntiles, 64 Msupertiles(128 rows), 3 matrices)
// ---------------------------------------------------------------------------
__global__ __launch_bounds__(256) void proj_gemm(
    const float* __restrict__ x,
    const float* __restrict__ Wq, const float* __restrict__ bq,
    const float* __restrict__ Wk, const float* __restrict__ bk,
    const float* __restrict__ Wv, const float* __restrict__ bv,
    float* __restrict__ qraw, float* __restrict__ kraw, float* __restrict__ vbuf)
{
    __shared__ __bf16 wtile[16 * DPAD];   // ~24.3 KB, column-major (n-major)

    const int lane  = threadIdx.x & 31;
    const int wave  = threadIdx.x >> 5;
    const int Nbase = blockIdx.x * 16;
    const int Mbase = blockIdx.y * 128 + wave * 16;

    const float* W;  const float* bias;  float* out;
    if      (blockIdx.z == 0) { W = Wq; bias = bq; out = qraw; }
    else if (blockIdx.z == 1) { W = Wk; bias = bk; out = kraw; }
    else                      { W = Wv; bias = bv; out = vbuf; }

    // Stage W panel: wtile[n][k] = bf16(W[k][Nbase+n]); global reads coalesced.
    for (int i = threadIdx.x; i < D_ * 16; i += 256) {
        const int k = i >> 4, n = i & 15;
        wtile[n * DPAD + k] = (__bf16)W[(size_t)k * D_ + Nbase + n];
    }
    __syncthreads();

    // A row for this lane (16-bit A 16x32 layout, ISA 7.12.2)
    const int mrow = Mbase + (lane & 15);
    const int b    = mrow >> 10;
    const int p    = mrow & 1023;
    const float* arow = x + (size_t)(b * (P_ + 1) + p + 1) * D_;  // skip CLS row
    const int aoff = (lane < 16) ? 0 : 8;    // A: K 0-7 / 8-15 then 16-23 / 24-31
    const int boff = (lane < 16) ? 0 : 8;    // B: K 0-7 / 8-15 then 16-23 / 24-31
    const int ncol = Nbase + (lane & 15);
    const __bf16* wcol = &wtile[(lane & 15) * DPAD];

    v8f acc = {};
    for (int k0 = 0; k0 < D_; k0 += 32) {
        const float4 a0 = *(const float4*)(arow + k0 + aoff);
        const float4 a1 = *(const float4*)(arow + k0 + aoff + 4);
        const float4 a2 = *(const float4*)(arow + k0 + 16 + aoff);
        const float4 a3 = *(const float4*)(arow + k0 + 16 + aoff + 4);
        v16bf a;
        a[0]  = (__bf16)a0.x; a[1]  = (__bf16)a0.y; a[2]  = (__bf16)a0.z; a[3]  = (__bf16)a0.w;
        a[4]  = (__bf16)a1.x; a[5]  = (__bf16)a1.y; a[6]  = (__bf16)a1.z; a[7]  = (__bf16)a1.w;
        a[8]  = (__bf16)a2.x; a[9]  = (__bf16)a2.y; a[10] = (__bf16)a2.z; a[11] = (__bf16)a2.w;
        a[12] = (__bf16)a3.x; a[13] = (__bf16)a3.y; a[14] = (__bf16)a3.z; a[15] = (__bf16)a3.w;

        const v8bf w0 = *(const v8bf*)(wcol + k0 + boff);
        const v8bf w1 = *(const v8bf*)(wcol + k0 + 16 + boff);
        v16bf bm;
        #pragma unroll
        for (int j = 0; j < 8; ++j) { bm[j] = w0[j]; bm[j + 8] = w1[j]; }

        acc = __builtin_amdgcn_wmma_f32_16x16x32_bf16(
                  false, a, false, bm, (short)0, acc, false, false);
    }

    // C/D layout: VGPR r -> M = r + (lane<16 ? 0 : 8), N = lane&15
    const int  mst = Mbase + ((lane < 16) ? 0 : 8);
    const float bb = bias[ncol];
    #pragma unroll
    for (int r = 0; r < 8; ++r)
        out[(size_t)(mst + r) * D_ + ncol] = acc[r] + bb;
}

// ---------------------------------------------------------------------------
// Kernel 2: row-wise L2-normalize q and k, convert to bf16.
// ---------------------------------------------------------------------------
__global__ __launch_bounds__(256) void l2norm_cvt(
    const float* __restrict__ qraw, const float* __restrict__ kraw,
    __bf16* __restrict__ qn, __bf16* __restrict__ kn)
{
    const int row = blockIdx.x;
    const int t   = threadIdx.x;
    __shared__ float sred[8];

    for (int which = 0; which < 2; ++which) {
        const float* src = which ? kraw : qraw;
        __bf16*      dst = which ? kn   : qn;
        const size_t base = (size_t)row * D_;
        float v0 = src[base + t];
        float v1 = src[base + t + 256];
        float v2 = src[base + t + 512];
        float s  = v0 * v0 + v1 * v1 + v2 * v2;
        #pragma unroll
        for (int o = 16; o > 0; o >>= 1) s += __shfl_xor(s, o, 32);
        if ((t & 31) == 0) sred[t >> 5] = s;
        __syncthreads();
        float tot = 0.f;
        #pragma unroll
        for (int i = 0; i < 8; ++i) tot += sred[i];
        const float sc = 1.0f / fmaxf(sqrtf(tot), 1e-12f);
        dst[base + t]       = (__bf16)(v0 * sc);
        dst[base + t + 256] = (__bf16)(v1 * sc);
        dst[base + t + 512] = (__bf16)(v2 * sc);
        __syncthreads();
    }
}

// ---------------------------------------------------------------------------
// Kernel 3: scores = qn@kn^T + bias, diag mask, top-16, softmax, gather v.
// One block per (batch, 16-row p-tile). 256 threads = 8 waves.
// q-tile staged via GLOBAL_LOAD_ASYNC_TO_LDS_B128 (ASYNCcnt) when available.
// ---------------------------------------------------------------------------
__global__ __launch_bounds__(256) void route_attend(
    const __bf16* __restrict__ qn, const __bf16* __restrict__ kn,
    const float* __restrict__ vbuf, const float* __restrict__ pos_bias,
    float* __restrict__ out)
{
    __shared__ __bf16 qtile[16 * D_];   // 24 KB
    __shared__ float  scores[16 * P_];  // 64 KB

    const int lane  = threadIdx.x & 31;
    const int wave  = threadIdx.x >> 5;
    const int b     = blockIdx.y;
    const int pbase = blockIdx.x * 16;

    // Stage q tile into LDS (contiguous 24 KB copy)
    {
        const __bf16* gsrc = qn + (size_t)(b * P_ + pbase) * D_;
#ifdef HAVE_ASYNC
        // 1536 x 16B chunks; each thread issues 6 async b128 copies.
        for (int i = threadIdx.x; i < (16 * D_) / 8; i += 256)
            __builtin_amdgcn_global_load_async_to_lds_b128(
                GLB_V4(gsrc) + i, LDS_V4(qtile) + i, 0, 0);
        asm volatile("s_wait_asynccnt 0" ::: "memory");
#else
        for (int i = threadIdx.x; i < 16 * D_ / 8; i += 256)
            *(v8bf*)(qtile + i * 8) = *(const v8bf*)(gsrc + i * 8);
#endif
    }
    __syncthreads();

    // Score GEMM: 64 column tiles of 16 k-positions, round-robin over waves.
    const int aoff = (lane < 16) ? 0 : 8;
    const int boff = (lane < 16) ? 0 : 8;
    const int arow = lane & 15;
    for (int ct = wave; ct < P_ / 16; ct += 8) {
        const int cbase = ct * 16;
        const __bf16* krow = kn + (size_t)(b * P_ + cbase + (lane & 15)) * D_;
        v8f acc = {};
        for (int k0 = 0; k0 < D_; k0 += 32) {
            const v8bf qa0 = *(const v8bf*)(qtile + arow * D_ + k0 + aoff);
            const v8bf qa1 = *(const v8bf*)(qtile + arow * D_ + k0 + 16 + aoff);
            const v8bf kb0 = *(const v8bf*)(krow + k0 + boff);
            const v8bf kb1 = *(const v8bf*)(krow + k0 + 16 + boff);
            v16bf a, bm;
            #pragma unroll
            for (int j = 0; j < 8; ++j) {
                a[j] = qa0[j];  a[j + 8] = qa1[j];
                bm[j] = kb0[j]; bm[j + 8] = kb1[j];
            }
            acc = __builtin_amdgcn_wmma_f32_16x16x32_bf16(
                      false, a, false, bm, (short)0, acc, false, false);
        }
        const int col = cbase + (lane & 15);
        const int m0  = (lane < 16) ? 0 : 8;
        #pragma unroll
        for (int r = 0; r < 8; ++r) {
            const int m    = m0 + r;
            const int prow = pbase + m;
            float s = acc[r] + pos_bias[(size_t)prow * P_ + col];
            if (prow == col) s = NEGV;
            scores[m * P_ + col] = s;
        }
    }
    __syncthreads();

    // Top-16 + softmax + gather-weighted-sum. Two rows per wave.
    for (int rr = 0; rr < 2; ++rr) {
        const int m = wave * 2 + rr;
        float* srow = &scores[m * P_];
        float topv[K_];
        int   topi[K_];
        #pragma unroll
        for (int it = 0; it < K_; ++it) {
            float best = -3.4e38f;
            int   bidx = P_;
            for (int i = lane; i < P_; i += 32) {
                float v = srow[i];
                if (v > best) { best = v; bidx = i; }
            }
            #pragma unroll
            for (int o = 16; o > 0; o >>= 1) {
                float ov = __shfl_xor(best, o, 32);
                int   oi = __shfl_xor(bidx, o, 32);
                if (ov > best || (ov == best && oi < bidx)) { best = ov; bidx = oi; }
            }
            topv[it] = best;
            topi[it] = bidx;
            if (lane == 0) srow[bidx] = -3.4e38f;  // same-wave LDS is in-order
        }

        // softmax over top-16 with 1/TEMP = 10
        const float mx = topv[0];
        float wgt[K_], wsum = 0.f;
        #pragma unroll
        for (int i = 0; i < K_; ++i) {
            wgt[i] = __expf((topv[i] - mx) * 10.0f);
            wsum  += wgt[i];
        }
        const float inv = 1.0f / wsum;

        // out[b, pbase+m, :] = sum_k w_k * v[b, route_k, :]
        float accd[24];
        #pragma unroll
        for (int j = 0; j < 24; ++j) accd[j] = 0.f;
        #pragma unroll
        for (int i = 0; i < K_; ++i) {
            const float w   = wgt[i] * inv;
            const float* vr = vbuf + (size_t)(b * P_ + topi[i]) * D_;
            #pragma unroll
            for (int j = 0; j < 24; ++j)
                accd[j] += w * vr[lane + j * 32];
        }
        float* orow = out + (size_t)(b * P_ + pbase + m) * D_;
        #pragma unroll
        for (int j = 0; j < 24; ++j)
            orow[lane + j * 32] = accd[j];
    }
}

// ---------------------------------------------------------------------------
extern "C" void kernel_launch(void* const* d_in, const int* in_sizes, int n_in,
                              void* d_out, int out_size, void* d_ws, size_t ws_size,
                              hipStream_t stream)
{
    const float* x        = (const float*)d_in[0];
    const float* Wq       = (const float*)d_in[1];
    const float* bq       = (const float*)d_in[2];
    const float* Wk       = (const float*)d_in[3];
    const float* bk       = (const float*)d_in[4];
    const float* Wv       = (const float*)d_in[5];
    const float* bv       = (const float*)d_in[6];
    const float* pos_bias = (const float*)d_in[7];
    float* outp = (float*)d_out;

    const size_t NROW = (size_t)B_ * P_;      // 8192
    float*  qraw = (float*)d_ws;
    float*  kraw = qraw + NROW * D_;
    float*  vbuf = kraw + NROW * D_;
    __bf16* qnb  = (__bf16*)(vbuf + NROW * D_);
    __bf16* knb  = qnb + NROW * D_;

    // 1) projections: 3 GEMMs (8192x768 @ 768x768)
    proj_gemm<<<dim3(D_ / 16, (int)(NROW / 128), 3), 256, 0, stream>>>(
        x, Wq, bq, Wk, bk, Wv, bv, qraw, kraw, vbuf);

    // 2) L2-normalize q,k rows -> bf16
    l2norm_cvt<<<dim3((int)NROW), 256, 0, stream>>>(qraw, kraw, qnb, knb);

    // 3) score GEMM + bias + mask + top-k + softmax + gather
    route_attend<<<dim3(P_ / 16, B_), 256, 0, stream>>>(
        qnb, knb, vbuf, pos_bias, outp);
}
